// EWMA_model2_56882546868815
// MI455X (gfx1250) — matrix operand order repaired and checked by reading, not compiled, not run on hardware
//
#include <hip/hip_runtime.h>
#include <cmath>

typedef __attribute__((ext_vector_type(16))) _Float16 v16h;
typedef __attribute__((ext_vector_type(8)))  _Float16 v8h;
typedef __attribute__((ext_vector_type(8)))  float    v8f;

#define WIN 20
#define P 256
#define CH 96          // K-chunk held in LDS (multiple of 32)
#define NTHREADS 1024

// workspace layout (floats)
#define WS_W      0
#define WS_SQRTW  1024
#define WS_INVDEN 2048
#define WS_RESE   4096
#define WS_RESS   8192

// ---------------------------------------------------------------------------
// Kernel 1: EWMA weights w_t = a^(nt-1-t) normalized to sum nt, sqrt(w), and
// 1/denom where denom = nt*(1 - sum(w^2)/nt^2).
// ---------------------------------------------------------------------------
__global__ void __launch_bounds__(1024) prep_kernel(const float* __restrict__ a,
                                                    const int* __restrict__ lagp,
                                                    float* __restrict__ ws) {
  __shared__ float red[1024];
  const int tid = threadIdx.x;
  const int nt  = lagp[0] * WIN;
  const float av = a[0];

  float s = 0.f;
  for (int t = tid; t < nt; t += 1024) {
    float wr = powf(av, (float)(nt - 1 - t));
    ws[WS_W + t] = wr;
    s += wr;
  }
  red[tid] = s;
  __syncthreads();
  for (int off = 512; off > 0; off >>= 1) {
    if (tid < off) red[tid] += red[tid + off];
    __syncthreads();
  }
  const float scale = (float)nt / red[0];
  __syncthreads();

  float s2 = 0.f;
  for (int t = tid; t < nt; t += 1024) {
    float wn = ws[WS_W + t] * scale;
    ws[WS_W + t]     = wn;
    ws[WS_SQRTW + t] = sqrtf(fabsf(wn));
    s2 += wn * wn;
  }
  red[tid] = s2;
  __syncthreads();
  for (int off = 512; off > 0; off >>= 1) {
    if (tid < off) red[tid] += red[tid + off];
    __syncthreads();
  }
  if (tid == 0) {
    float denom = (float)nt * (1.f - red[0] / ((float)nt * (float)nt));
    ws[WS_INVDEN] = 1.f / denom;
  }
}

// ---------------------------------------------------------------------------
// Kernel 2: one workgroup (32 wave32s) per backtest window.
//   Phase 1: weighted mean (coalesced partials through S scratch)
//   Phase 2: Sigma = Z^T Z via v_wmma_f32_16x16x32_f16, Z = sqrt(w)*(Y-mean)
//            in LDS, t-contiguous (Z[p][t]) so fragments are ds_load_b128s.
//            All 18 fragment loads of a K-step issued before the 8 WMMAs so
//            the scheduler can use partial s_wait_dscnt and overlap DS with
//            the matrix pipe.
//   Phase 3: Gauss-Jordan solve Sigma y = 1 in LDS
//   Phase 4: w_opt = p*y/sum(y); realized test-window returns
// ---------------------------------------------------------------------------
__global__ void __launch_bounds__(1024, 1) gmv_kernel(const float* __restrict__ Y,
                                                      const int* __restrict__ lagp,
                                                      float* __restrict__ ws,
                                                      int nrows) {
  __shared__ float    S[P * P];          // 256 KB: Sigma (and mean scratch)
  __shared__ _Float16 Zs[P * CH];        // 48 KB: sqrt(w)-scaled centered chunk
  __shared__ float    meanS[P];
  __shared__ float    bS[P];
  __shared__ float    pivS;
  __shared__ float    scal;
  __shared__ float    r20[WIN];

  const int lag   = lagp[0];
  const int nt    = lag * WIN;
  const int n_win = nrows / WIN - lag;
  const int w     = blockIdx.x;
  if (w >= n_win) return;

  const int tid  = threadIdx.x;
  const int lane = tid & 31;
  const int wv   = tid >> 5;             // wave id 0..31
  const int srow = w * WIN;              // first training row

  const float* wsW  = ws + WS_W;
  const float* wsSq = ws + WS_SQRTW;

  // ---- phase 1: weighted mean ----
  {
    const int p = tid & (P - 1);
    const int q = tid >> 8;              // 0..3
    float part = 0.f;
    for (int t = q; t < nt; t += 4)
      part += wsW[t] * Y[(size_t)(srow + t) * P + p];
    S[q * P + p] = part;
  }
  __syncthreads();
  if (tid < P)
    meanS[tid] = (S[tid] + S[P + tid] + S[2 * P + tid] + S[3 * P + tid]) / (float)nt;

  // ---- phase 2: Gram matrix via WMMA ----
  // wave wv owns output tile row i = wv/2, tile cols j0..j0+7
  const int i   = wv >> 1;
  const int j0  = (wv & 1) * 8;
  const int m   = lane & 15;
  const int lhi = (lane >> 4) & 1;

  v8f acc[8];
  #pragma unroll
  for (int t = 0; t < 8; ++t)
    #pragma unroll
    for (int e = 0; e < 8; ++e) acc[t][e] = 0.f;

  const int nchunks = (nt + CH - 1) / CH;
  for (int cidx = 0; cidx < nchunks; ++cidx) {
    const int cbase = cidx * CH;
    __syncthreads();
    {   // fill Z chunk: Z[p][tl] = (Y - mean) * sqrt(w), zero-padded past nt
      const int p = tid & (P - 1);
      for (int tl = tid >> 8; tl < CH; tl += 4) {
        const int t = cbase + tl;
        float z = 0.f;
        if (t < nt) {
          z = (Y[(size_t)(srow + t) * P + p] - meanS[p]) * wsSq[t];
          // warm L2/L0 for the next chunk while this one is being consumed
          if (t + CH < nt)
            __builtin_prefetch(&Y[(size_t)(srow + t + CH) * P + p], 0, 3);
        }
        Zs[p * CH + tl] = (_Float16)z;
      }
    }
    __syncthreads();
    #pragma unroll
    for (int ks = 0; ks < CH; ks += 32) {
      // A fragment (16x32 f16): lanes 0-15 hold K {0..7,16..23}, lanes 16-31 {8..15,24..31}
      const _Float16* za = Zs + (i * 16 + m) * CH + ks + lhi * 8;
      v8h alo = *(const v8h*)(za);
      v8h ahi = *(const v8h*)(za + 16);
      v16h A = __builtin_shufflevector(alo, ahi,
                 0,1,2,3,4,5,6,7,8,9,10,11,12,13,14,15);
      // issue all 8 B fragment loads before any WMMA consumes them
      v16h Bf[8];
      #pragma unroll
      for (int tj = 0; tj < 8; ++tj) {
        // B fragment (32x16 f16): lanes 0-15 hold K {0..7,16..23}, lanes 16-31 {8..15,24..31}
        const _Float16* zb = Zs + ((j0 + tj) * 16 + m) * CH + ks + lhi * 8;
        v8h blo = *(const v8h*)(zb);
        v8h bhi = *(const v8h*)(zb + 16);
        Bf[tj] = __builtin_shufflevector(blo, bhi,
                   0,1,2,3,4,5,6,7,8,9,10,11,12,13,14,15);
      }
      #pragma unroll
      for (int tj = 0; tj < 8; ++tj) {
        acc[tj] = __builtin_amdgcn_wmma_f32_16x16x32_f16(
            /*neg_a=*/false, A, /*neg_b=*/false, Bf[tj],
            /*c_mod=*/(short)0, acc[tj], /*reuse_a=*/false, /*reuse_b=*/false);
      }
    }
  }
  __syncthreads();

  // write Sigma = acc / denom into LDS (C/D layout: vgpr r -> M=r(+8), lane -> N)
  {
    const float invden = ws[WS_INVDEN];
    const int rowbase = i * 16 + lhi * 8;
    const int colin   = lane & 15;
    #pragma unroll
    for (int tj = 0; tj < 8; ++tj) {
      const int col = (j0 + tj) * 16 + colin;
      #pragma unroll
      for (int r = 0; r < 8; ++r)
        S[(rowbase + r) * P + col] = acc[tj][r] * invden;
    }
  }
  if (tid < P) bS[tid] = 1.f;
  __syncthreads();

  // ---- phase 3: Gauss-Jordan solve S y = 1 (SPD, no pivoting) ----
  for (int k = 0; k < P; ++k) {
    if (tid == 0) pivS = 1.f / S[k * P + k];
    __syncthreads();
    const float piv = pivS;
    for (int jc = k + 1 + tid; jc < P; jc += NTHREADS)
      S[k * P + jc] *= piv;
    if (tid == NTHREADS - 1) bS[k] *= piv;
    __syncthreads();
    const float bk  = bS[k];
    const int irow  = tid >> 2;
    const int sub   = tid & 3;
    if (irow != k) {
      const float f = S[irow * P + k];
      for (int jc = k + 1 + sub; jc < P; jc += 4)
        S[irow * P + jc] -= f * S[k * P + jc];
      if (sub == 0) bS[irow] -= f * bk;
    }
    __syncthreads();
  }

  // ---- phase 4: w_opt and realized test-window returns ----
  if (tid == 0) {
    float sum = 0.f;
    for (int p2 = 0; p2 < P; ++p2) sum += bS[p2];
    scal = sum;
  }
  __syncthreads();
  if (tid < P) meanS[tid] = (float)P * bS[tid] / scal;   // reuse meanS as w_opt
  __syncthreads();
  if (tid < WIN) {
    const float* yr = Y + (size_t)(srow + nt + tid) * P;
    float accr = 0.f;
    for (int p2 = 0; p2 < P; ++p2) accr += yr[p2] * meanS[p2];
    r20[tid] = accr;
  }
  __syncthreads();
  if (tid == 0) {
    float se = 0.f;
    for (int t = 0; t < WIN; ++t) se += r20[t];
    const float mu = se / (float)WIN;
    float var = 0.f;
    for (int t = 0; t < WIN; ++t) { float d = r20[t] - mu; var += d * d; }
    var /= (float)(WIN - 1);
    ws[WS_RESE + w] = se;
    ws[WS_RESS + w] = var;
  }
}

// ---------------------------------------------------------------------------
// Kernel 3: reduce per-window results to (vol, mu, sharpe)
// ---------------------------------------------------------------------------
__global__ void __launch_bounds__(256) fin_kernel(const int* __restrict__ lagp,
                                                  const float* __restrict__ ws,
                                                  float* __restrict__ out,
                                                  int nrows) {
  __shared__ float re[256], rs[256];
  const int tid   = threadIdx.x;
  const int n_win = nrows / WIN - lagp[0];
  float a = 0.f, b = 0.f;
  for (int w = tid; w < n_win; w += 256) {
    a += ws[WS_RESE + w];
    b += ws[WS_RESS + w];
  }
  re[tid] = a; rs[tid] = b;
  __syncthreads();
  for (int off = 128; off > 0; off >>= 1) {
    if (tid < off) { re[tid] += re[tid + off]; rs[tid] += rs[tid + off]; }
    __syncthreads();
  }
  if (tid == 0) {
    const float me  = re[0] / (float)n_win;
    const float ms  = rs[0] / (float)n_win;
    const float vol = sqrtf(ms * 252.f);
    const float mu  = me / (float)WIN * 252.f;
    out[0] = vol;
    out[1] = mu;
    out[2] = mu / vol;
  }
}

// ---------------------------------------------------------------------------
extern "C" void kernel_launch(void* const* d_in, const int* in_sizes, int n_in,
                              void* d_out, int out_size, void* d_ws, size_t ws_size,
                              hipStream_t stream) {
  const float* Y    = (const float*)d_in[0];
  const float* a    = (const float*)d_in[1];
  // d_in[2] = b (unused by the 'S' estimator branch)
  const int*   lagp = (const int*)d_in[3];
  float* ws  = (float*)d_ws;
  float* out = (float*)d_out;

  const int nrows  = in_sizes[0] / P;   // 20000
  const int maxwin = nrows / WIN;       // upper bound on n_win (blocks self-guard)

  prep_kernel<<<1, 1024, 0, stream>>>(a, lagp, ws);
  gmv_kernel<<<maxwin, 1024, 0, stream>>>(Y, lagp, ws, nrows);
  fin_kernel<<<1, 256, 0, stream>>>(lagp, ws, out, nrows);
}
